// Encoder_Predictor_79731772883065
// MI455X (gfx1250) — matrix operand order repaired and checked by reading, not compiled, not run on hardware
//
#include <hip/hip_runtime.h>
#include <cstddef>

// ---------------------------------------------------------------------------
// MI455X (gfx1250) implementation of the tree-GRU encoder/predictor.
// Shapes fixed by setup_inputs: B=64, L=512, D_IN=256, H=256, A=128, DEPTH=4
//
//  * All GEMMs on v_wmma_f32_16x16x32_bf16 (wave32, fp32 accum).
//  * Weights packed once into exact ISA B-fragment order -> one contiguous
//    32B load per lane per fragment (L2-resident).
//  * Activations kept in bf16 in memory/LDS: A-fragments are two contiguous
//    16B loads, conversions happen once per produced value (v_cvt_pk_bf16).
//  * xp GEMM stages its A-slab via the Tensor Data Mover (tensor_load_to_lds
//    + s_wait_tensorcnt) -- one 32KB DMA instead of 8x redundant reads.
//  * Recurrence = persistent 1-workgroup kernel (32 wave32s on one WGP),
//    h carried in registers (f32) + bf16 LDS mirror; next-step xp/xt slabs
//    prefetched (global_prefetch_b8) to hide latency behind WMMA phases.
// ---------------------------------------------------------------------------

typedef __attribute__((ext_vector_type(16))) __bf16 v16bf;
typedef __attribute__((ext_vector_type(8)))  __bf16 v8bf;
typedef __attribute__((ext_vector_type(8)))  float  v8f;
typedef __attribute__((ext_vector_type(4)))  unsigned int u32x4;
typedef __attribute__((ext_vector_type(8)))  int i32x8;
typedef __attribute__((ext_vector_type(4)))  int i32x4;

#define Bk   64
#define Lk   512
#define Hk   256
#define Ak   128
#define H3k  768
#define DEPTHk 4

union FragBF { v16bf v; v8bf h2[2]; __bf16 h[16]; };

// A fragment (16x32 bf16) from a bf16 row-major source (global or LDS).
// ISA layout: lane -> row m=lane&15, half=lane>>4;
//   elems 0..7  : K = kc*32 + 8*half + e
//   elems 8..15 : K = kc*32 + 16 + 8*half + e
__device__ __forceinline__ v16bf load_a_frag_bf(const __bf16* base, int ld,
                                                int row0, int kc, int lane) {
  const int m = lane & 15, half = lane >> 4;
  const __bf16* p = base + (size_t)(row0 + m) * ld + kc * 32 + 8 * half;
  FragBF f;
  f.h2[0] = *reinterpret_cast<const v8bf*>(p);
  f.h2[1] = *reinterpret_cast<const v8bf*>(p + 16);
  return f.v;
}

// A fragment from an f32 source with row gather (embedding GEMM only).
__device__ __forceinline__ v16bf load_a_frag_embed(const float* x, int r0,
                                                   int kc, int lane) {
  const int m = lane & 15, half = lane >> 4;
  const int r = r0 + m;                       // r = l*64 + b
  const float* p = x + ((size_t)(r & 63) * Lk + (r >> 6)) * 256 + kc * 32;
  FragBF f;
#pragma unroll
  for (int e = 0; e < 8; ++e) {
    f.h[e]     = (__bf16)p[8 * half + e];
    f.h[8 + e] = (__bf16)p[16 + 8 * half + e];
  }
  return f.v;
}

// B fragment: pre-packed, one contiguous 32-byte load per lane.
__device__ __forceinline__ v16bf load_b_frag(const __bf16* pack, int ntiles,
                                             int kc, int nt, int lane) {
  return *reinterpret_cast<const v16bf*>(
      pack + ((((size_t)kc * ntiles + nt) * 32 + lane) << 4));
}

__device__ __forceinline__ v8f wmma_bf16(v16bf a, v16bf b, v8f c) {
  return __builtin_amdgcn_wmma_f32_16x16x32_bf16(false, a, false, b,
                                                 (short)0, c, false, false);
}

__device__ __forceinline__ v8f zero8() {
  v8f z;
#pragma unroll
  for (int i = 0; i < 8; ++i) z[i] = 0.f;
  return z;
}

__device__ __forceinline__ void store_tile(float* out, int ldo, int row0,
                                           int col0, v8f acc, int lane) {
  const int n = lane & 15, half = lane >> 4;
#pragma unroll
  for (int v = 0; v < 8; ++v)
    out[(size_t)(row0 + v + 8 * half) * ldo + col0 + n] = acc[v];
}

__device__ __forceinline__ float sigmoidf_(float x) {
  return 1.f / (1.f + expf(-x));
}

// TDM: DMA a 2-D bf16 tile (rows x 256 elems, row stride 256) from global
// memory into LDS offset 0.  D# per CDNA5 ISA ch.8; issued by one wave.
__device__ __forceinline__ void tdm_load_rows_bf16(unsigned long long gaddr,
                                                   unsigned rows) {
  // group0: [1:0]=count=1 | lds_addr=0 | global_addr[56:0] | type=2
  u32x4 g0 = { 1u, 0u, (unsigned)(gaddr & 0xffffffffu),
               (unsigned)(((gaddr >> 32) & 0x01ffffffu) | 0x80000000u) };
  // group1: data_size=1(2B); tensor_dim0=256; tensor_dim1=rows;
  //         tile_dim0=256; tile_dim1=rows; tensor_dim0_stride=256
  i32x8 g1 = { (int)(1u << 16),          // workgroup_mask=0 | data_size=1
               (int)(256u << 16),        // tensor_dim0 lo16 (hi bits -> next)
               (int)(rows << 16),        // tensor_dim0 hi=0 | tensor_dim1 lo16
               (int)(256u << 16),        // tensor_dim1 hi=0 | tile_dim0
               (int)rows,                // tile_dim1 | tile_dim2=0
               256, 0, 0 };              // tensor_dim0_stride=256; dim1_stride=0
  i32x4 g2 = { 0, 0, 0, 0 };
  i32x4 g3 = { 0, 0, 0, 0 };
#if __clang_major__ >= 23
  i32x8 g4 = { 0, 0, 0, 0, 0, 0, 0, 0 };
  __builtin_amdgcn_tensor_load_to_lds(g0, g1, g2, g3, g4, 0);
#else
  __builtin_amdgcn_tensor_load_to_lds(g0, g1, g2, g3, 0);
#endif
  __builtin_amdgcn_s_wait_tensorcnt(0);
}

// ---------------------------------------------------------------------------
// Pack an f32 weight (K=256 rows x N cols, row-major) into WMMA B-fragment
// order.  One-time cost.
// ---------------------------------------------------------------------------
__global__ __launch_bounds__(256) void pack_weight_kernel(
    const float* __restrict__ src, __bf16* __restrict__ dst,
    int N, int total) {
  int tid = blockIdx.x * 256 + threadIdx.x;
  if (tid >= total) return;
  const int ntiles = N >> 4;
  const int e  = tid & 15;
  const int l  = (tid >> 4) & 31;
  const int rest = tid >> 9;
  const int nt = rest % ntiles;
  const int kc = rest / ntiles;
  const int Kr = kc * 32 + ((l >> 4) << 3) + (e & 7) + ((e >> 3) << 4);
  const int Nc = (nt << 4) + (l & 15);
  dst[tid] = (__bf16)src[(size_t)Kr * N + Nc];
}

// mask (B,L) bool -> time-major int (L,B)
__global__ __launch_bounds__(256) void mask_init_kernel(
    const unsigned char* __restrict__ mask_in, int* __restrict__ maskbuf) {
  int i = blockIdx.x * 256 + threadIdx.x;
  if (i >= Lk * Bk) return;
  const int b = i & 63, t = i >> 6;
  maskbuf[i] = mask_in[(size_t)b * Lk + t] ? 1 : 0;
}

// hseq_bf(l,b,:) = bf16( x[b,l,:] @ W_emb + b_emb )  (M=32768, N=256, K=256)
__global__ __launch_bounds__(256) void embed_kernel(
    const float* __restrict__ x, const __bf16* __restrict__ wpack,
    const float* __restrict__ bias, __bf16* __restrict__ hseq_bf) {
  const int w = threadIdx.x >> 5, lane = threadIdx.x & 31;
  const int rBase = blockIdx.x * 64;
  const int nt = blockIdx.y * 8 + w;          // global 16-col tile, 0..15
  const int n = lane & 15, half = lane >> 4;
  const float bv = bias[nt * 16 + n];
#pragma unroll
  for (int rt = 0; rt < 4; ++rt) {
    v8f acc = zero8();
#pragma unroll
    for (int kc = 0; kc < 8; ++kc) {
      v16bf a = load_a_frag_embed(x, rBase + rt * 16, kc, lane);
      v16bf b = load_b_frag(wpack, 16, kc, nt, lane);
      acc = wmma_bf16(a, b, acc);
    }
#pragma unroll
    for (int v = 0; v < 8; ++v)
      hseq_bf[(size_t)(rBase + rt * 16 + v + 8 * half) * Hk + nt * 16 + n] =
          (__bf16)(acc[v] + bv);
  }
}

// xp_raw = h_seq @ W  (M=32768, N=768, K=256).  A-slab (64x256 bf16, 32KB)
// staged in LDS by the Tensor Data Mover, then shared by all 8 waves.
__global__ __launch_bounds__(256) void xp_gemm_kernel(
    const __bf16* __restrict__ hseq_bf, const __bf16* __restrict__ wpack,
    float* __restrict__ xp) {
  extern __shared__ char smemraw[];
  __bf16* As = (__bf16*)smemraw;              // 64 x 256 bf16
  const int w = threadIdx.x >> 5, lane = threadIdx.x & 31;
  const int rBase = blockIdx.x * 64;
  const int nt = blockIdx.y * 8 + w;          // 0..47

  if (threadIdx.x < 32) {                     // one wave issues the DMA
    tdm_load_rows_bf16(
        (unsigned long long)(const void*)(hseq_bf + (size_t)rBase * Hk), 64u);
  }
  __syncthreads();

#pragma unroll
  for (int rt = 0; rt < 4; ++rt) {
    v8f acc = zero8();
#pragma unroll
    for (int kc = 0; kc < 8; ++kc) {
      v16bf a = load_a_frag_bf(As, Hk, rt * 16, kc, lane);
      v16bf b = load_b_frag(wpack, 48, kc, nt, lane);
      acc = wmma_bf16(a, b, acc);
    }
    store_tile(xp, H3k, rBase + rt * 16, nt * 16, acc, lane);
  }
}

// In-place LayerNorm of each 768-wide row of xp with gamma[0]/beta[0].
__global__ __launch_bounds__(256) void ln_rows_kernel(
    float* __restrict__ xp, const float* __restrict__ g,
    const float* __restrict__ bta) {
  __shared__ float s1[256], s2[256];
  const int r = blockIdx.x, tid = threadIdx.x;
  float* row = xp + (size_t)r * H3k;
  const float a0 = row[tid], a1 = row[tid + 256], a2 = row[tid + 512];
  s1[tid] = a0 + a1 + a2;
  s2[tid] = a0 * a0 + a1 * a1 + a2 * a2;
  __syncthreads();
  for (int s = 128; s > 0; s >>= 1) {
    if (tid < s) { s1[tid] += s1[tid + s]; s2[tid] += s2[tid + s]; }
    __syncthreads();
  }
  const float mean = s1[0] * (1.f / H3k);
  const float var  = s2[0] * (1.f / H3k) - mean * mean;
  const float rstd = rsqrtf(var + 1e-5f);
  row[tid]       = g[tid]       * (a0 - mean) * rstd + bta[tid];
  row[tid + 256] = g[tid + 256] * (a1 - mean) * rstd + bta[tid + 256];
  row[tid + 512] = g[tid + 512] * (a2 - mean) * rstd + bta[tid + 512];
}

// ---------------------------------------------------------------------------
// Persistent recurrence: one workgroup (32 wave32s), 512 time steps.
// LDS: hu/hp (64x768 f32, 192KB) | t1 (64x128 f32, 32KB) |
//      logits (128 f32) | hbf (64x256 bf16, 32KB)  => 262,656 bytes.
// ---------------------------------------------------------------------------
__global__ __launch_bounds__(1024, 1) void scan_kernel(
    const float* __restrict__ xp, float* __restrict__ hseq,
    __bf16* __restrict__ hseq_bf, int* __restrict__ maskbuf,
    const __bf16* __restrict__ Upack,
    const __bf16* __restrict__ Wa1p,
    const __bf16* __restrict__ Ua1p,
    const float* __restrict__ bvec,   // b (3H)
    const float* __restrict__ g1, const float* __restrict__ b1,
    const float* __restrict__ ba1,    // (A)
    const float* __restrict__ Wa2,    // (A,2) f32
    const float* __restrict__ ba2,    // (2)
    float* __restrict__ out_action, float* __restrict__ out_policy, int d) {
  extern __shared__ char smem[];
  float*  hu  = (float*)smem;                    // 64*768 f32  (192KB)
  float*  t1  = (float*)(smem + 196608);         // 64*128 f32  ( 32KB)
  float*  lg  = (float*)(smem + 229376);         // 128 f32
  __bf16* hbf = (__bf16*)(smem + 229888);        // 64*256 bf16 ( 32KB)

  const int tid = threadIdx.x, w = tid >> 5, lane = tid & 31;
  const int n = lane & 15, half = lane >> 4;

  float hreg[16];                                // h0 = zeros
#pragma unroll
  for (int k = 0; k < 16; ++k) hreg[k] = 0.f;
  for (int i = tid; i < Bk * Hk; i += 1024) hbf[i] = (__bf16)0.f;
  __syncthreads();

  for (int t = 0; t < Lk; ++t) {
    // Prefetch next step's gate inputs (192KB xp slab) and xt slab (32KB)
    // so they sit in cache by the time Phase C / next Phase A needs them.
    if (t + 1 < Lk) {
      const float* xpn = xp + ((size_t)(t + 1) * Bk) * H3k;
      for (int pi = tid; pi < 1536; pi += 1024)
        __builtin_prefetch(xpn + (size_t)pi * 32, 0, 0);      // 128B/line
      const __bf16* xtn = hseq_bf + (size_t)(t + 1) * (Bk * Hk);
      if (tid < 256)
        __builtin_prefetch(xtn + (size_t)tid * 64, 0, 0);
    }

    // ---- Phase A: hu = h @ U (64x768) ; t1 = tanh(xt@Wa1 + h@Ua1 + ba1) ----
    for (int i = 0; i < 6; ++i) {                 // 192 U-tiles over 32 waves
      const int tt = w + i * 32;
      const int rt = tt / 48, nt = tt % 48;
      v8f acc = zero8();
#pragma unroll
      for (int kc = 0; kc < 8; ++kc) {
        v16bf a = load_a_frag_bf(hbf, Hk, rt * 16, kc, lane);
        v16bf b = load_b_frag(Upack, 48, kc, nt, lane);
        acc = wmma_bf16(a, b, acc);
      }
      store_tile(hu, H3k, rt * 16, nt * 16, acc, lane);
    }
    {                                             // 32 action tiles, 1/wave
      const int rt = w >> 3, nt = w & 7;
      const __bf16* xtbase = hseq_bf + (size_t)t * (Bk * Hk);
      v8f acc = zero8();
#pragma unroll
      for (int kc = 0; kc < 8; ++kc) {
        v16bf a1f = load_a_frag_bf(xtbase, Hk, rt * 16, kc, lane);
        v16bf b1f = load_b_frag(Wa1p, 8, kc, nt, lane);
        acc = wmma_bf16(a1f, b1f, acc);
        v16bf a2f = load_a_frag_bf(hbf, Hk, rt * 16, kc, lane);
        v16bf b2f = load_b_frag(Ua1p, 8, kc, nt, lane);
        acc = wmma_bf16(a2f, b2f, acc);
      }
      const int col = nt * 16 + n;
      const float bb = ba1[col];
#pragma unroll
      for (int v = 0; v < 8; ++v)
        t1[(size_t)(rt * 16 + v + 8 * half) * Ak + col] = tanhf(acc[v] + bb);
    }
    __syncthreads();

    // ---- Phase B: hp = LN(hu; g1, b1), in place. 2 rows per wave. ----
    for (int rr = 0; rr < 2; ++rr) {
      const int row = w * 2 + rr;
      float s = 0.f, s2 = 0.f;
      for (int i = lane; i < H3k; i += 32) {
        const float v = hu[(size_t)row * H3k + i];
        s += v; s2 += v * v;
      }
      for (int off = 16; off > 0; off >>= 1) {
        s  += __shfl_xor(s, off, 32);
        s2 += __shfl_xor(s2, off, 32);
      }
      const float mean = s * (1.f / H3k);
      const float var  = s2 * (1.f / H3k) - mean * mean;
      const float rstd = rsqrtf(var + 1e-5f);
      for (int i = lane; i < H3k; i += 32) {
        const size_t ix = (size_t)row * H3k + i;
        hu[ix] = g1[i] * (hu[ix] - mean) * rstd + b1[i];
      }
    }
    __syncthreads();

    // ---- Phase C: GRU gates (h in registers) + policy logits (tid<128) ----
#pragma unroll
    for (int k = 0; k < 16; ++k) {
      const int i = tid + (k << 10);
      const int b_ = i >> 8, j = i & 255;
      const size_t xpb = ((size_t)t * Bk + b_) * H3k;
      const size_t hb  = (size_t)b_ * H3k;
      const float r  = sigmoidf_(xp[xpb + j]       + hu[hb + j]       + bvec[j]);
      const float z  = sigmoidf_(xp[xpb + 256 + j] + hu[hb + 256 + j] + bvec[256 + j]);
      const float hh = tanhf   (xp[xpb + 512 + j] + r * hu[hb + 512 + j] + bvec[512 + j]);
      const float hprev = hreg[k];
      const float hn = z * hprev + (1.f - z) * hh;
      const float ho = maskbuf[t * Bk + b_] ? hn : hprev;
      hreg[k] = ho;
      hbf[i] = (__bf16)ho;                         // feeds next step's WMMA
      hseq[(size_t)t * (Bk * Hk) + i]    = ho;     // f32 (final output)
      hseq_bf[(size_t)t * (Bk * Hk) + i] = (__bf16)ho;  // feeds next depth
    }
    if (tid < 128) {                               // logits = t1 @ Wa2 + ba2
      const int b_ = tid >> 1, c = tid & 1;
      float acc = ba2[c];
      for (int k = 0; k < Ak; ++k) acc += t1[b_ * Ak + k] * Wa2[k * 2 + c];
      lg[tid] = acc;
    }
    __syncthreads();

    // ---- Phase D: softmax, action, outputs, mask update ----
    if (tid < Bk) {
      const float l0 = lg[tid * 2], l1 = lg[tid * 2 + 1];
      const float mx = fmaxf(l0, l1);
      const float e0 = expf(l0 - mx), e1 = expf(l1 - mx);
      const float inv = 1.f / (e0 + e1);
      const float p0 = e0 * inv, p1 = e1 * inv;
      const int mt = maskbuf[t * Bk + tid];
      const int act = (p1 > p0) && mt;
      const size_t base = ((size_t)tid * DEPTHk + d) * Lk + t;
      out_action[base]         = act ? 1.f : 0.f;
      out_policy[base * 2]     = p0;
      out_policy[base * 2 + 1] = p1;
      maskbuf[t * Bk + tid] = act;                 // m_seq & act for next depth
    }
    __syncthreads();
  }
}

__global__ __launch_bounds__(256) void final_copy_kernel(
    const float* __restrict__ hseq, float* __restrict__ out) {
  const int i = blockIdx.x * 256 + threadIdx.x;
  if (i < Bk * Hk) out[i] = hseq[(size_t)(Lk - 1) * Bk * Hk + i];
}

// ---------------------------------------------------------------------------
extern "C" void kernel_launch(void* const* d_in, const int* in_sizes, int n_in,
                              void* d_out, int out_size, void* d_ws,
                              size_t ws_size, hipStream_t stream) {
  (void)in_sizes; (void)n_in; (void)out_size; (void)ws_size;
  const float* x      = (const float*)d_in[0];
  const unsigned char* mask = (const unsigned char*)d_in[1];  // jax bool
  const float* W_emb  = (const float*)d_in[2];
  const float* b_emb  = (const float*)d_in[3];
  const float* W      = (const float*)d_in[4];
  const float* U      = (const float*)d_in[5];
  const float* bvec   = (const float*)d_in[6];
  const float* Wa1    = (const float*)d_in[7];
  const float* Ua1    = (const float*)d_in[8];
  const float* ba1    = (const float*)d_in[9];
  const float* Wa2    = (const float*)d_in[10];
  const float* ba2    = (const float*)d_in[11];
  const float* gammas = (const float*)d_in[12];
  const float* betas  = (const float*)d_in[13];

  // workspace layout (bytes)
  char* ws = (char*)d_ws;
  float*  hseq    = (float*) (ws);                    //  33,554,432
  float*  xp      = (float*) (ws + 33554432);         // 100,663,296
  int*    maskbf  = (int*)   (ws + 134217728);        //     131,072
  __bf16* hseq_bf = (__bf16*)(ws + 134348800);        //  16,777,216
  __bf16* WembP   = (__bf16*)(ws + 151126016);        //     131,072
  __bf16* WP      = (__bf16*)(ws + 151257088);        //     393,216
  __bf16* UP      = (__bf16*)(ws + 151650304);        //     393,216
  __bf16* Wa1P    = (__bf16*)(ws + 152043520);        //      65,536
  __bf16* Ua1P    = (__bf16*)(ws + 152109056);        //      65,536

  float* out        = (float*)d_out;
  float* out_action = out + 16384;                    // (B,DEPTH,L)
  float* out_policy = out + 16384 + 131072;           // (B,DEPTH,L,2)

  const int SMEM = 262656;                            // scan dynamic LDS
  hipFuncSetAttribute(reinterpret_cast<const void*>(scan_kernel),
                      hipFuncAttributeMaxDynamicSharedMemorySize, SMEM);

  // one-time weight packs (deterministic; re-run every call)
  pack_weight_kernel<<<256, 256, 0, stream>>>(W_emb, WembP, 256, 65536);
  pack_weight_kernel<<<768, 256, 0, stream>>>(W,     WP,    768, 196608);
  pack_weight_kernel<<<768, 256, 0, stream>>>(U,     UP,    768, 196608);
  pack_weight_kernel<<<128, 256, 0, stream>>>(Wa1,   Wa1P,  128, 32768);
  pack_weight_kernel<<<128, 256, 0, stream>>>(Ua1,   Ua1P,  128, 32768);
  mask_init_kernel<<<128, 256, 0, stream>>>(mask, maskbf);

  embed_kernel<<<dim3(512, 2), 256, 0, stream>>>(x, WembP, b_emb, hseq_bf);

  for (int d = 0; d < DEPTHk; ++d) {
    xp_gemm_kernel<<<dim3(512, 6), 256, 32768, stream>>>(hseq_bf, WP, xp);
    ln_rows_kernel<<<32768, 256, 0, stream>>>(xp, gammas, betas);
    scan_kernel<<<1, 1024, SMEM, stream>>>(
        xp, hseq, hseq_bf, maskbf, UP, Wa1P, Ua1P, bvec,
        gammas + H3k, betas + H3k, ba1, Wa2, ba2,
        out_action, out_policy, d);
  }
  final_copy_kernel<<<64, 256, 0, stream>>>(hseq, out);
}
